// DAGTaskEncoder_68642167324672
// MI455X (gfx1250) — matrix operand order repaired and checked by reading, not compiled, not run on hardware
//
#include <hip/hip_runtime.h>
#include <hip/hip_bf16.h>

#define NNODES 100000
#define NEDGES 1600000
#define NGRAPH 64
#define MPAD   100096   // NNODES rounded up to multiple of 128

typedef __attribute__((ext_vector_type(16))) __bf16 v16bf;
typedef __attribute__((ext_vector_type(8)))  __bf16 v8bf;
typedef __attribute__((ext_vector_type(8)))  float  v8f;

// ---------------- helpers ----------------

__device__ __forceinline__ unsigned short f2bf(float f) {
  union { float f; unsigned u; } t; t.f = f;
  unsigned r = t.u + 0x7FFFu + ((t.u >> 16) & 1u);   // round-to-nearest-even
  return (unsigned short)(r >> 16);
}

__device__ __forceinline__ void atomicMaxF(float* addr, float val) {
  // sign-aware int trick; memory must be initialized to -inf (0xFF800000)
  if (val >= 0.0f) atomicMax((int*)addr, __float_as_int(val));
  else             atomicMin((unsigned int*)addr, __float_as_uint(val));
}

__global__ void fill_u32_kernel(unsigned* __restrict__ p, unsigned v, int n) {
  int i = blockIdx.x * blockDim.x + threadIdx.x;
  if (i < n) p[i] = v;
}

// fp32 activations -> bf16 bits, zero-padding rows [nValid, MPAD)
__global__ void cvt_act_bf16_kernel(const float* __restrict__ x, unsigned short* __restrict__ xb,
                                    int nValid, int D) {
  int idx = blockIdx.x * blockDim.x + threadIdx.x;
  if (idx >= MPAD * D) return;
  int r = idx / D;
  xb[idx] = (r < nValid) ? f2bf(x[idx]) : (unsigned short)0;
}

// W[K,Nc] fp32 (row-major) -> Wt[Nc,K] bf16 bits (transposed, for contiguous B-fragment loads)
__global__ void cvt_w_kernel(const float* __restrict__ W, unsigned short* __restrict__ Wt,
                             int K, int Nc) {
  int idx = blockIdx.x * blockDim.x + threadIdx.x;
  if (idx >= K * Nc) return;
  int k = idx / Nc, c = idx - k * Nc;
  Wt[(size_t)c * K + k] = f2bf(W[idx]);
}

// ---------------- WMMA GEMM: C[M,Nc] = A[Mp,K](bf16) @ Wt[Nc,K](bf16)^T + bias ----------------
// block = 256 threads = 8 waves; wave w -> rows [bx*128 + w*16, +16), cols [by*16, +16); K multiple of 32.
__global__ void gemm_wmma_bias_kernel(const unsigned short* __restrict__ A,
                                      const unsigned short* __restrict__ Wt,
                                      const float* __restrict__ bias,
                                      float* __restrict__ C,
                                      int M, int K, int Nc) {
  int wv   = threadIdx.x >> 5;
  int lane = threadIdx.x & 31;
  int rowBase = blockIdx.x * 128 + wv * 16;
  int colBase = blockIdx.y * 16;
  int half = lane >> 4;     // lane[4]
  int rl   = lane & 15;

  const unsigned short* arow = A  + (size_t)(rowBase + rl) * K;  // A lanes 0-15/16-31 both hold M=0..15
  const unsigned short* bcol = Wt + (size_t)(colBase + rl) * K;  // B lane holds one column

  v8f acc = {};
  for (int k0 = 0; k0 < K; k0 += 32) {
    // A 16x32 bf16: elems 0..7 -> K = k0 + half*8 + e ; elems 8..15 -> K = k0 + 16 + half*8 + e
    v8bf a0 = *(const v8bf*)(arow + k0 + half * 8);
    v8bf a1 = *(const v8bf*)(arow + k0 + 16 + half * 8);
    // B 32x16 bf16: elems 0..7 -> K = k0 + half*8 + e ; elems 8..15 -> K = k0 + 16 + half*8 + e
    v8bf b0 = *(const v8bf*)(bcol + k0 + half * 8);
    v8bf b1 = *(const v8bf*)(bcol + k0 + 16 + half * 8);
    v16bf a, b;
#pragma unroll
    for (int i = 0; i < 8; ++i) { a[i] = a0[i]; a[i + 8] = a1[i]; b[i] = b0[i]; b[i + 8] = b1[i]; }
    acc = __builtin_amdgcn_wmma_f32_16x16x32_bf16(false, a, false, b, (short)0, acc, false, false);
  }
  int col = colBase + rl;
  float bv = bias[col];
#pragma unroll
  for (int i = 0; i < 8; ++i) {           // C/D: row = i + half*8, col = rl
    int row = rowBase + half * 8 + i;
    if (row < M) C[(size_t)row * Nc + col] = acc[i] + bv;
  }
}

// ---------------- edge pass 1: logits + segment-max (wave per edge) ----------------
template<int H>
__global__ void edge_logits_kernel(const float* __restrict__ xl, const float* __restrict__ xr,
                                   const float* __restrict__ att, const int* __restrict__ ei,
                                   float* __restrict__ logits, float* __restrict__ m, int E_) {
  const int D = H * 64, EL = D / 32;
  int wv = (blockIdx.x * blockDim.x + threadIdx.x) >> 5;
  int lane = threadIdx.x & 31;
  if (wv >= E_) return;
  int src = ei[wv], dst = ei[E_ + wv];
  const float* pl = xl + (size_t)src * D;
  const float* pr = xr + (size_t)dst * D;
  int i0 = lane * EL;
  float a[4], b[4], c[4];
  if (EL == 4) {
    float4 t = *(const float4*)(pl + i0); a[0]=t.x; a[1]=t.y; a[2]=t.z; a[3]=t.w;
    float4 u = *(const float4*)(pr + i0); b[0]=u.x; b[1]=u.y; b[2]=u.z; b[3]=u.w;
    float4 v = *(const float4*)(att + i0); c[0]=v.x; c[1]=v.y; c[2]=v.z; c[3]=v.w;
  } else {
    float2 t = *(const float2*)(pl + i0); a[0]=t.x; a[1]=t.y;
    float2 u = *(const float2*)(pr + i0); b[0]=u.x; b[1]=u.y;
    float2 v = *(const float2*)(att + i0); c[0]=v.x; c[1]=v.y;
  }
  float partial = 0.0f;
#pragma unroll
  for (int j = 0; j < EL; ++j) {
    float e = a[j] + b[j];
    e = e > 0.0f ? e : 0.2f * e;            // LeakyReLU(0.2)
    partial += e * c[j];
  }
  const int span = (H == 2) ? 8 : 16;       // H==2: per-head reduce in each half-wave
#pragma unroll
  for (int o = 1; o <= span; o <<= 1) partial += __shfl_xor(partial, o, 32);
  if ((lane & (2 * span - 1)) == 0) {
    int h = (H == 2) ? (lane >> 4) : 0;
    logits[(size_t)wv * H + h] = partial;
    atomicMaxF(&m[(size_t)dst * H + h], partial);
  }
}

// ---------------- edge pass 2: p = exp(logit - m[dst]); denom += p ----------------
template<int H>
__global__ void edge_exp_kernel(float* __restrict__ logits, const float* __restrict__ m,
                                float* __restrict__ den, const int* __restrict__ ei, int E_) {
  int idx = blockIdx.x * blockDim.x + threadIdx.x;
  if (idx >= E_ * H) return;
  int e = idx / H, h = idx - e * H;
  int dst = ei[E_ + e];
  float p = __expf(logits[idx] - m[(size_t)dst * H + h]);
  logits[idx] = p;                           // overwrite logits with p
  atomicAdd(&den[(size_t)dst * H + h], p);
}

// ---------------- edge pass 3: out[dst] += alpha * xl[src] (wave per edge) ----------------
template<int H>
__global__ void edge_agg_kernel(const float* __restrict__ xl, const float* __restrict__ p,
                                const float* __restrict__ den, const int* __restrict__ ei,
                                float* __restrict__ outAcc, int E_) {
  const int D = H * 64, EL = D / 32;
  int wv = (blockIdx.x * blockDim.x + threadIdx.x) >> 5;
  int lane = threadIdx.x & 31;
  if (wv >= E_) return;
  int src = ei[wv], dst = ei[E_ + wv];
  int i0 = lane * EL;
  int h = (H == 2) ? (lane >> 4) : 0;
  float alpha = p[(size_t)wv * H + h] / (den[(size_t)dst * H + h] + 1e-16f);
  const float* pl = xl + (size_t)src * D;
  float* po = outAcc + (size_t)dst * D;
  float a[4];
  if (EL == 4) { float4 t = *(const float4*)(pl + i0); a[0]=t.x; a[1]=t.y; a[2]=t.z; a[3]=t.w; }
  else         { float2 t = *(const float2*)(pl + i0); a[0]=t.x; a[1]=t.y; }
#pragma unroll
  for (int j = 0; j < EL; ++j) atomicAdd(&po[i0 + j], alpha * a[j]);
}

// ---------------- bias1 + LayerNorm + ELU -> bf16 h (wave per row, zero-pad rows) ----------------
__global__ void finalize1_kernel(const float* __restrict__ out1, const float* __restrict__ bias,
                                 const float* __restrict__ g, const float* __restrict__ be,
                                 unsigned short* __restrict__ hb, int N_) {
  int wv = (blockIdx.x * blockDim.x + threadIdx.x) >> 5;
  int lane = threadIdx.x & 31;
  if (wv >= MPAD) return;
  int i0 = lane * 4;
  unsigned short* dst = hb + (size_t)wv * 128 + i0;
  if (wv >= N_) { *(unsigned long long*)dst = 0ull; return; }
  float4 t = *(const float4*)(out1 + (size_t)wv * 128 + i0);
  float v0 = t.x + bias[i0], v1 = t.y + bias[i0 + 1], v2 = t.z + bias[i0 + 2], v3 = t.w + bias[i0 + 3];
  float s  = v0 + v1 + v2 + v3;
  float sq = v0 * v0 + v1 * v1 + v2 * v2 + v3 * v3;
#pragma unroll
  for (int o = 1; o < 32; o <<= 1) { s += __shfl_xor(s, o, 32); sq += __shfl_xor(sq, o, 32); }
  float mu = s * (1.0f / 128.0f);
  float var = sq * (1.0f / 128.0f) - mu * mu;
  float rs = rsqrtf(fmaxf(var, 0.0f) + 1e-5f);
  float y[4] = { v0, v1, v2, v3 };
  unsigned long long pk = 0;
#pragma unroll
  for (int j = 0; j < 4; ++j) {
    float z = (y[j] - mu) * rs * g[i0 + j] + be[i0 + j];
    z = z > 0.0f ? z : (__expf(z) - 1.0f);            // ELU
    pk |= (unsigned long long)f2bf(z) << (16 * j);
  }
  *(unsigned long long*)dst = pk;
}

// ---------------- node_emb = out2 + bias2 -> d_out; accumulate per-graph sums ----------------
__global__ void finalize2_kernel(const float* __restrict__ out2, const float* __restrict__ bias2,
                                 const int* __restrict__ batch, float* __restrict__ nodeOut,
                                 float* __restrict__ gsum, int N_) {
  int idx = blockIdx.x * blockDim.x + threadIdx.x;
  if (idx >= N_ * 64) return;
  int n = idx >> 6, c = idx & 63;
  float v = out2[idx] + bias2[c];
  nodeOut[idx] = v;
  atomicAdd(&gsum[(batch[n] << 6) + c], v);
}

__global__ void count_kernel(const int* __restrict__ batch, float* __restrict__ gcnt, int N_) {
  int i = blockIdx.x * blockDim.x + threadIdx.x;
  if (i < N_) atomicAdd(&gcnt[batch[i]], 1.0f);
}

__global__ void pool_finish_kernel(const float* __restrict__ gsum, const float* __restrict__ gcnt,
                                   float* __restrict__ out, int G_) {
  int idx = blockIdx.x * blockDim.x + threadIdx.x;
  if (idx >= G_ * 64) return;
  out[idx] = gsum[idx] / fmaxf(gcnt[idx >> 6], 1.0f);
}

// ---------------- host ----------------

extern "C" void kernel_launch(void* const* d_in, const int* in_sizes, int n_in,
                              void* d_out, int out_size, void* d_ws, size_t ws_size,
                              hipStream_t stream) {
  (void)in_sizes; (void)n_in; (void)out_size; (void)ws_size;
  const float* x     = (const float*)d_in[0];
  const int*   ei    = (const int*)d_in[1];
  const int*   batch = (const int*)d_in[2];
  const float* W1l   = (const float*)d_in[3];
  const float* b1l   = (const float*)d_in[4];
  const float* W1r   = (const float*)d_in[5];
  const float* b1r   = (const float*)d_in[6];
  const float* att1  = (const float*)d_in[7];
  const float* bias1 = (const float*)d_in[8];
  const float* g1    = (const float*)d_in[9];
  const float* be1   = (const float*)d_in[10];
  const float* W2l   = (const float*)d_in[11];
  const float* b2l   = (const float*)d_in[12];
  const float* W2r   = (const float*)d_in[13];
  const float* b2r   = (const float*)d_in[14];
  const float* att2  = (const float*)d_in[15];
  const float* bias2 = (const float*)d_in[16];
  float* out = (float*)d_out;

  const int N = NNODES, E = NEDGES, G = NGRAPH, Mp = MPAD;

  char* w = (char*)d_ws; size_t off = 0;
  auto take = [&](size_t bytes) -> void* {
    void* p = w + off; off = (off + bytes + 255) & ~(size_t)255; return p;
  };
  unsigned short* x_bf = (unsigned short*)take((size_t)Mp * 128 * 2); // reused as h_bf after layer 1
  unsigned short* w1lt = (unsigned short*)take(128 * 128 * 2);
  unsigned short* w1rt = (unsigned short*)take(128 * 128 * 2);
  unsigned short* w2lt = (unsigned short*)take(64 * 128 * 2);
  unsigned short* w2rt = (unsigned short*)take(64 * 128 * 2);
  float* xl1     = (float*)take((size_t)N * 128 * 4);   // reused as out2 after layer-1 aggregation
  float* xr1     = (float*)take((size_t)N * 128 * 4);   // reused as xl2 / xr2
  float* out1    = (float*)take((size_t)N * 128 * 4);
  float* logits1 = (float*)take((size_t)E * 2 * 4);     // reused as logits2
  float* m1      = (float*)take((size_t)N * 2 * 4);
  float* den1    = (float*)take((size_t)N * 2 * 4);
  float* m2      = (float*)take((size_t)N * 4);
  float* den2    = (float*)take((size_t)N * 4);
  float* gsum    = (float*)take((size_t)G * 64 * 4);
  float* gcnt    = (float*)take((size_t)G * 4);

  unsigned short* h_bf = x_bf;
  float* xl2 = xr1;
  float* xr2 = xr1 + (size_t)N * 64;
  float* out2 = xl1;
  float* logits2 = logits1;

  // per-call init (harness does not re-poison between replays)
  hipMemsetAsync(den1, 0, (size_t)N * 2 * 4, stream);
  hipMemsetAsync(den2, 0, (size_t)N * 4, stream);
  hipMemsetAsync(out1, 0, (size_t)N * 128 * 4, stream);
  hipMemsetAsync(gsum, 0, (size_t)G * 64 * 4, stream);
  hipMemsetAsync(gcnt, 0, (size_t)G * 4, stream);
  fill_u32_kernel<<<(N * 2 + 255) / 256, 256, 0, stream>>>((unsigned*)m1, 0xFF800000u, N * 2);
  fill_u32_kernel<<<(N + 255) / 256, 256, 0, stream>>>((unsigned*)m2, 0xFF800000u, N);

  // precision prep: bf16 copies (x padded; weights transposed)
  cvt_act_bf16_kernel<<<(Mp * 128 + 255) / 256, 256, 0, stream>>>(x, x_bf, N, 128);
  cvt_w_kernel<<<(128 * 128 + 255) / 256, 256, 0, stream>>>(W1l, w1lt, 128, 128);
  cvt_w_kernel<<<(128 * 128 + 255) / 256, 256, 0, stream>>>(W1r, w1rt, 128, 128);
  cvt_w_kernel<<<(128 * 64 + 255) / 256, 256, 0, stream>>>(W2l, w2lt, 128, 64);
  cvt_w_kernel<<<(128 * 64 + 255) / 256, 256, 0, stream>>>(W2r, w2rt, 128, 64);

  // layer 1: projections via WMMA
  dim3 gd1(Mp / 128, 128 / 16);
  gemm_wmma_bias_kernel<<<gd1, 256, 0, stream>>>(x_bf, w1lt, b1l, xl1, N, 128, 128);
  gemm_wmma_bias_kernel<<<gd1, 256, 0, stream>>>(x_bf, w1rt, b1r, xr1, N, 128, 128);

  // layer 1: edge passes (wave per edge, 8 edges per block)
  int eb = (E + 7) / 8;
  edge_logits_kernel<2><<<eb, 256, 0, stream>>>(xl1, xr1, att1, ei, logits1, m1, E);
  edge_exp_kernel<2><<<(E * 2 + 255) / 256, 256, 0, stream>>>(logits1, m1, den1, ei, E);
  edge_agg_kernel<2><<<eb, 256, 0, stream>>>(xl1, logits1, den1, ei, out1, E);

  // bias + LayerNorm + ELU -> bf16 h (with zero row padding)
  finalize1_kernel<<<(Mp * 32 + 255) / 256, 256, 0, stream>>>(out1, bias1, g1, be1, h_bf, N);

  // layer 2
  hipMemsetAsync(out2, 0, (size_t)N * 64 * 4, stream);   // xl1 is dead now; safe to reuse
  dim3 gd2(Mp / 128, 64 / 16);
  gemm_wmma_bias_kernel<<<gd2, 256, 0, stream>>>(h_bf, w2lt, b2l, xl2, N, 128, 64);
  gemm_wmma_bias_kernel<<<gd2, 256, 0, stream>>>(h_bf, w2rt, b2r, xr2, N, 128, 64);

  edge_logits_kernel<1><<<eb, 256, 0, stream>>>(xl2, xr2, att2, ei, logits2, m2, E);
  edge_exp_kernel<1><<<(E + 255) / 256, 256, 0, stream>>>(logits2, m2, den2, ei, E);
  edge_agg_kernel<1><<<eb, 256, 0, stream>>>(xl2, logits2, den2, ei, out2, E);

  // node_emb -> d_out[0:N*64], then mean pooling -> d_out[N*64:]
  finalize2_kernel<<<(N * 64 + 255) / 256, 256, 0, stream>>>(out2, bias2, batch, out, gsum, N);
  count_kernel<<<(N + 255) / 256, 256, 0, stream>>>(batch, gcnt, N);
  pool_finish_kernel<<<(G * 64 + 255) / 256, 256, 0, stream>>>(gsum, gcnt, out + (size_t)N * 64, G);
}
